// Qwen3NextGatedDeltaNet_6640019440432
// MI455X (gfx1250) — matrix operand order repaired and checked by reading, not compiled, not run on hardware
//
#include <hip/hip_runtime.h>
#include <hip/hip_bf16.h>

// ---------------------------------------------------------------------------
// Qwen3-Next GatedDeltaNet layer for MI455X (gfx1250, wave32, WMMA + TDM)
// All contractions run on v_wmma_f32_16x16x32_bf16 (fp32 accumulate).
// Core kernel stages q/k/v chunk tiles into LDS via the Tensor Data Mover.
// ---------------------------------------------------------------------------

typedef __attribute__((ext_vector_type(16))) __bf16 v16bf;
typedef __attribute__((ext_vector_type(8)))  float  v8f;

typedef unsigned int tdm_v4u __attribute__((ext_vector_type(4)));
typedef int          tdm_v8i __attribute__((ext_vector_type(8)));
typedef int          tdm_v4i __attribute__((ext_vector_type(4)));

#if defined(__has_builtin)
#if __has_builtin(__builtin_amdgcn_tensor_load_to_lds)
#define HAVE_TDM 1
#endif
#endif
#ifndef HAVE_TDM
#define HAVE_TDM 0
#endif

#define S_LEN 4096
#define EMB   2048
#define NKH   16
#define NVH   32
#define DK    128
#define DV    128
#define CHK   64
#define NCH   (S_LEN / CHK)
#define CONV_DIM 8192
#define QKVZ_COLS 12288

union Frag { v16bf v; __bf16 b[16]; };
union Acc8 { v8f  v; float  f[8]; };

// native convert: gfx1250 has hardware f32->bf16 converts
static __device__ __forceinline__ __bf16 f2bf(float x) { return (__bf16)x; }

static __device__ __forceinline__ v8f wmma_bf16(v16bf a, v16bf b, v8f c) {
  return __builtin_amdgcn_wmma_f32_16x16x32_bf16(false, a, false, b, (short)0, c, false, false);
}

static __device__ __forceinline__ void zero_acc(Acc8& a) {
#pragma unroll
  for (int i = 0; i < 8; ++i) a.f[i] = 0.f;
}

// ---- WMMA fragment loaders (contiguous k order, same in A and B, wave32) -----------------
// A 16x32 (MxK): lanes 0-15 rows M; element e -> k = half*16 + e
// B 32x16 (KxN): lane = column; lanes 0-15 K 0..15, lanes 16-31 K 16..31.

static __device__ __forceinline__ v16bf ld_a_rm(const __bf16* p, int lda, int m0, int k0) {
  const int lane = threadIdx.x & 31;
  const int half = lane >> 4;
  const __bf16* row = p + (m0 + (lane & 15)) * lda + k0;
  Frag f;
#pragma unroll
  for (int e = 0; e < 16; ++e) {
    const int k = half * 16 + e;
    f.b[e] = row[k];
  }
  return f.v;
}

// A[m][k] stored column-major: element at p[k*ldm + m]
static __device__ __forceinline__ v16bf ld_a_cm(const __bf16* p, int ldm, int m0, int k0) {
  const int lane = threadIdx.x & 31;
  const int half = lane >> 4;
  const int m = m0 + (lane & 15);
  Frag f;
#pragma unroll
  for (int e = 0; e < 16; ++e) {
    const int k = half * 16 + e;
    f.b[e] = p[(k0 + k) * ldm + m];
  }
  return f.v;
}

// B stored as B^T: p[n][k] with contiguous k (fast path -> ds_load_b128)
static __device__ __forceinline__ v16bf ld_b_cm(const __bf16* p, int ldb, int n0, int k0) {
  const int lane = threadIdx.x & 31;
  const int half = lane >> 4;
  const __bf16* col = p + (n0 + (lane & 15)) * ldb + k0 + half * 16;
  Frag f;
#pragma unroll
  for (int e = 0; e < 16; ++e) f.b[e] = col[e];
  return f.v;
}

// B row-major p[k][n] (strided gather)
static __device__ __forceinline__ v16bf ld_b_rm(const __bf16* p, int ldn, int n0, int k0) {
  const int lane = threadIdx.x & 31;
  const int half = lane >> 4;
  const int n = n0 + (lane & 15);
  const int kb = k0 + half * 16;
  Frag f;
#pragma unroll
  for (int e = 0; e < 16; ++e) f.b[e] = p[(kb + e) * ldn + n];
  return f.v;
}

// B row-major from f32 LDS (convert on the fly) — used against the fp32 state
static __device__ __forceinline__ v16bf ld_b_rm_f32(const float* p, int ldn, int n0, int k0) {
  const int lane = threadIdx.x & 31;
  const int half = lane >> 4;
  const int n = n0 + (lane & 15);
  const int kb = k0 + half * 16;
  Frag f;
#pragma unroll
  for (int e = 0; e < 16; ++e) f.b[e] = f2bf(p[(kb + e) * ldn + n]);
  return f.v;
}

// C/D 16x16 f32 layout: VGPR r -> M = r (lanes 0-15) / 8+r (lanes 16-31)
static __device__ __forceinline__ void st_tile(float* p, int ld, int m0, int n0, const Acc8& a) {
  const int lane = threadIdx.x & 31;
  const int n = n0 + (lane & 15);
  const int mb = m0 + ((lane >> 4) << 3);
#pragma unroll
  for (int r = 0; r < 8; ++r) p[(mb + r) * ld + n] = a.f[r];
}

static __device__ __forceinline__ void st_tile_bf16(__bf16* p, int ld, int m0, int n0, const Acc8& a) {
  const int lane = threadIdx.x & 31;
  const int n = n0 + (lane & 15);
  const int mb = m0 + ((lane >> 4) << 3);
#pragma unroll
  for (int r = 0; r < 8; ++r) p[(mb + r) * ld + n] = f2bf(a.f[r]);
}

#if HAVE_TDM
// ---------------------------------------------------------------------------
// Tensor Data Mover: 2D tile (rows x cols f32, row-major) global -> LDS.
// Hand-packed D# per CDNA5 ISA 8.3/8.4 (count=1, type=2/image, data_size=4B).
// 6-arg builtin (this toolchain): (v4u g0, v8i g1, v4i g2, v4i g3, v8i, cpol).
// Issue from one wave with full EXEC; TENSORcnt tracks completion.
// ---------------------------------------------------------------------------
static __device__ __forceinline__ void tdm_load_2d(unsigned lds_off, const void* gptr,
                                                   int rows, int cols,
                                                   int row_stride, int tensor_rows) {
  const unsigned long long ga = (unsigned long long)(size_t)gptr;
  tdm_v4u g0;
  g0[0] = 1u;                                         // count=1, user desc
  g0[1] = lds_off;                                    // lds_addr (bytes)
  g0[2] = (unsigned)(ga & 0xFFFFFFFFu);               // global_addr[31:0]
  g0[3] = (unsigned)((ga >> 32) & 0x01FFFFFFu)        // global_addr[56:32]
        | (2u << 30);                                 // type = 2 (image)
  tdm_v8i g1;
  g1[0] = (2 << 16);                                  // data_size = 4 bytes
  g1[1] = (int)(((unsigned)cols & 0xFFFFu) << 16);    // tensor_dim0[15:0]
  g1[2] = (int)((((unsigned)cols >> 16) & 0xFFFFu)    // tensor_dim0[31:16]
        | (((unsigned)tensor_rows & 0xFFFFu) << 16)); // tensor_dim1[15:0]
  g1[3] = (int)((((unsigned)tensor_rows >> 16) & 0xFFFFu)
        | (((unsigned)cols & 0xFFFFu) << 16));        // tile_dim0
  g1[4] = rows & 0xFFFF;                              // tile_dim1 (tile_dim2=0)
  g1[5] = row_stride;                                 // tensor_dim0_stride[31:0]
  g1[6] = 0;
  g1[7] = 0;
  tdm_v4i z4 = {0, 0, 0, 0};
  tdm_v8i z8 = {0, 0, 0, 0, 0, 0, 0, 0};
  __builtin_amdgcn_tensor_load_to_lds(g0, g1, z4, z4, z8, 0);
}
#endif

// ---------------------------------------------------------------------------
// Generic GEMM: C[M,N] = A[M,K] @ B[K,N], fp32 in/out, bf16 WMMA inside.
// Block tile 128x128x32, 256 threads (8 waves), wave tile 64x32 (4x2 WMMA).
// ---------------------------------------------------------------------------
#define BM 128
#define BN 128
#define BK 32

__global__ __launch_bounds__(256) void gemm_bf16wmma(const float* __restrict__ A,
                                                     const float* __restrict__ B,
                                                     float* __restrict__ C,
                                                     int M, int N, int K) {
  __shared__ __bf16 As[BM * BK];   // [m][k]
  __shared__ __bf16 Bs[BN * BK];   // B^T: [n][k]
  const int tid = threadIdx.x;
  const int lane = tid & 31, w = tid >> 5;
  const int bm = blockIdx.y * BM, bn = blockIdx.x * BN;
  const int wm = (w & 1) * 64, wn = (w >> 1) * 32;

  Acc8 acc[4][2];
#pragma unroll
  for (int i = 0; i < 4; ++i)
#pragma unroll
    for (int j = 0; j < 2; ++j) zero_acc(acc[i][j]);

  for (int kk = 0; kk < K; kk += BK) {
#pragma unroll
    for (int i = 0; i < 16; ++i) {            // A: 128x32 -> 16 elems/thread
      const int idx = tid * 16 + i;
      const int r = idx >> 5, k = idx & 31;
      As[r * BK + k] = f2bf(A[(size_t)(bm + r) * K + kk + k]);
    }
#pragma unroll
    for (int i = 0; i < 16; ++i) {            // B: 32x128 -> transpose into Bs
      const int idx = tid * 16 + i;
      const int k = idx >> 7, n = idx & 127;
      const float v = (bn + n < N) ? B[(size_t)(kk + k) * N + bn + n] : 0.f;
      Bs[n * BK + k] = f2bf(v);
    }
    if (kk + BK < K)                          // emits global_prefetch_b8
      __builtin_prefetch(&A[(size_t)(bm + (tid >> 1)) * K + kk + BK], 0, 1);
    __syncthreads();

    v16bf af[4], bf[2];
#pragma unroll
    for (int i = 0; i < 4; ++i) af[i] = ld_a_rm(As, BK, wm + 16 * i, 0);
#pragma unroll
    for (int j = 0; j < 2; ++j) bf[j] = ld_b_cm(Bs, BK, wn + 16 * j, 0);
#pragma unroll
    for (int i = 0; i < 4; ++i)
#pragma unroll
      for (int j = 0; j < 2; ++j) acc[i][j].v = wmma_bf16(af[i], bf[j], acc[i][j].v);
    __syncthreads();
  }

#pragma unroll
  for (int i = 0; i < 4; ++i)
#pragma unroll
    for (int j = 0; j < 2; ++j) {
      const int n = bn + wn + j * 16 + (lane & 15);
      if (n < N) {
        const int mb = bm + wm + i * 16 + ((lane >> 4) << 3);
#pragma unroll
        for (int r = 0; r < 8; ++r) C[(size_t)(mb + r) * N + n] = acc[i][j].f[r];
      }
    }
}

// ---------------------------------------------------------------------------
// Depthwise causal conv (K=4) + SiLU, scatter into head-major q/k/v buffers.
// ---------------------------------------------------------------------------
__global__ __launch_bounds__(256) void conv_silu_kernel(const float* __restrict__ qkvz,
                                                        const float* __restrict__ cw,
                                                        float* __restrict__ qc,
                                                        float* __restrict__ kc,
                                                        float* __restrict__ vc) {
  const int idx = blockIdx.x * 256 + threadIdx.x;     // t*8192 + c
  const int t = idx >> 13, c = idx & 8191;
  int col; float* outp;
  if (c < 2048) {                                     // q
    const int h = c >> 7, d = c & 127;
    col = h * 768 + d;
    outp = qc + (size_t)h * S_LEN * DK + (size_t)t * DK + d;
  } else if (c < 4096) {                              // k
    const int c2 = c - 2048, h = c2 >> 7, d = c2 & 127;
    col = h * 768 + 128 + d;
    outp = kc + (size_t)h * S_LEN * DK + (size_t)t * DK + d;
  } else {                                            // v
    const int c3 = c - 4096, h = c3 >> 7, d = c3 & 127;
    col = (h >> 1) * 768 + 256 + (h & 1) * 128 + d;
    outp = vc + (size_t)h * S_LEN * DV + (size_t)t * DV + d;
  }
  float acc = 0.f;
#pragma unroll
  for (int j = 0; j < 4; ++j) {
    const int ts = t - 3 + j;
    if (ts >= 0) acc += cw[j * CONV_DIM + c] * qkvz[(size_t)ts * QKVZ_COLS + col];
  }
  *outp = acc / (1.f + __expf(-acc));                 // SiLU
}

// ---------------------------------------------------------------------------
// Gates: beta = sigmoid(b), g = -exp(A_log) * softplus(a + dt_bias)
// ---------------------------------------------------------------------------
__global__ __launch_bounds__(256) void gbeta_kernel(const float* __restrict__ ba,
                                                    const float* __restrict__ A_log,
                                                    const float* __restrict__ dt_bias,
                                                    float* __restrict__ g,
                                                    float* __restrict__ beta) {
  const int idx = blockIdx.x * 256 + threadIdx.x;     // t*32 + h
  const int t = idx >> 5, h = idx & 31;
  const int kh = h >> 1, p = h & 1;
  const float bg = ba[(size_t)t * 64 + kh * 4 + p];
  const float ag = ba[(size_t)t * 64 + kh * 4 + 2 + p];
  beta[(size_t)h * S_LEN + t] = 1.f / (1.f + __expf(-bg));
  const float x = ag + dt_bias[h];
  const float sp = (x > 20.f) ? x : log1pf(__expf(x));
  g[(size_t)h * S_LEN + t] = -__expf(A_log[h]) * sp;
}

// ---------------------------------------------------------------------------
// Chunked gated delta-rule core. One workgroup per value-head; 64 sequential
// chunks; 128x128 fp32 state lives in LDS. q/k/v chunk tiles staged by TDM.
// ---------------------------------------------------------------------------
#define CORE_SMEM 311872
#define QS_OFF 114688u   // staging q  -> aliases vintf (32KB)
#define KS_OFF 147456u   // staging k  -> aliases tmpf  (32KB)
#define VS_OFF 65536u    // staging v  -> aliases Lf+attnf (32KB)

__global__ __launch_bounds__(256) void core_kernel(const float* __restrict__ qcg,
                                                   const float* __restrict__ kcg,
                                                   const float* __restrict__ vcg,
                                                   const float* __restrict__ gg,
                                                   const float* __restrict__ bb,
                                                   float* __restrict__ coreo) {
  extern __shared__ char sm[];
  float* state = (float*)sm;                    // 128*128 f32   (65536)
  float* Lf    = (float*)(sm + 65536);          // 64*64 f32
  float* attnf = (float*)(sm + 81920);          // 64*64 f32
  float* af    = (float*)(sm + 98304);          // 64*64 f32
  float* vintf = (float*)(sm + 114688);         // 64*128 f32
  float* tmpf  = (float*)(sm + 147456);         // 64*128 f32
  float* gcum  = (float*)(sm + 180224);         // 64
  float* betas = (float*)(sm + 180480);         // 64
  float* glast = (float*)(sm + 180736);         // 1 (padded)
  __bf16* qsb = (__bf16*)(sm + 180800);         // 64*128 each (16KB)
  __bf16* qgb = qsb + 8192;
  __bf16* kb  = qgb + 8192;
  __bf16* kbb = kb + 8192;
  __bf16* kgb = kbb + 8192;
  __bf16* ksb = kgb + 8192;
  __bf16* vbb = ksb + 8192;
  __bf16* attnb = vbb + 8192;                   // 64*64
  __bf16* ab  = attnb + 4096;                   // 64*64
  __bf16* vnewb = qsb;                          // alias: qsb dead after phase 3
  __bf16* kcdb  = kbb;                          // alias: kbb dead after phase 3

  const int tid = threadIdx.x;
  const int w = tid >> 5;
  const int h = blockIdx.x;
  const int khd = h >> 1;

  const float* qh = qcg + (size_t)khd * S_LEN * DK;
  const float* kh = kcg + (size_t)khd * S_LEN * DK;
  const float* vh = vcg + (size_t)h * S_LEN * DV;
  const float* gh = gg + (size_t)h * S_LEN;
  const float* bh = bb + (size_t)h * S_LEN;
  float* oh = coreo + (size_t)h * S_LEN * DV;

  for (int i = tid; i < DK * DV; i += 256) state[i] = 0.f;
  __syncthreads();

  for (int ch = 0; ch < NCH; ++ch) {
    const int t0 = ch * CHK;

#if HAVE_TDM
    // -- TDM: stage q/k/v 64x128 f32 chunk tiles into LDS (one op per wave)
    if (w == 0)      tdm_load_2d(QS_OFF, qh + (size_t)t0 * DK, CHK, DK, DK, S_LEN - t0);
    else if (w == 1) tdm_load_2d(KS_OFF, kh + (size_t)t0 * DK, CHK, DK, DK, S_LEN - t0);
    else if (w == 2) tdm_load_2d(VS_OFF, vh + (size_t)t0 * DV, CHK, DV, DV, S_LEN - t0);
    const float* qsrc = vintf;                  // QS_OFF
    const float* ksrc = tmpf;                   // KS_OFF
    const float* vsrc = Lf;                     // VS_OFF (spans Lf+attnf)
#else
    const float* qsrc = qh + (size_t)t0 * DK;
    const float* ksrc = kh + (size_t)t0 * DK;
    const float* vsrc = vh + (size_t)t0 * DV;
#endif

    // -- gates + prefix sum (overlaps with TDM transfer)
    if (tid < 64) { gcum[tid] = gh[t0 + tid]; betas[tid] = bh[t0 + tid]; }
    __syncthreads();
    if (tid == 0) {
      float s = 0.f;
      for (int i = 0; i < 64; ++i) { s += gcum[i]; gcum[i] = s; }
      glast[0] = s;
    }
#if HAVE_TDM
    if (w < 3) __builtin_amdgcn_s_wait_tensorcnt(0);
#endif
    __syncthreads();
    const float gl = glast[0];

    // -- build bf16 WMMA operands from staged chunk
    for (int i = tid; i < CHK * DK; i += 256) {
      const int t = i >> 7;
      const float gc = gcum[t], bt = betas[t];
      const float qv = qsrc[i] * 0.08838834764831845f;   // rsqrt(128)
      qsb[i] = f2bf(qv);
      qgb[i] = f2bf(qv * __expf(gc));
      const float kv = ksrc[i];
      kb[i]  = f2bf(kv);
      kbb[i] = f2bf(kv * bt);
      kgb[i] = f2bf(kv * bt * __expf(gc));
      ksb[i] = f2bf(kv * __expf(gl - gc));
      vbb[i] = f2bf(vsrc[i] * bt);
    }
    __syncthreads();

    // -- phase 3: Lraw = kbb@kb^T and araw = qsb@kb^T (32 tiles / 8 waves)
    for (int tt = w; tt < 32; tt += 8) {
      const int which = tt >> 4;
      const int t4 = tt & 15;
      const int m0 = (t4 >> 2) * 16, n0 = (t4 & 3) * 16;
      Acc8 acc; zero_acc(acc);
      const __bf16* Ap = which ? qsb : kbb;
#pragma unroll
      for (int ks = 0; ks < 4; ++ks) {
        v16bf a = ld_a_rm(Ap, DK, m0, ks * 32);
        v16bf b = ld_b_cm(kb, DK, n0, ks * 32);   // B = kb^T (k-contiguous)
        acc.v = wmma_bf16(a, b, acc.v);
      }
      st_tile(which ? af : Lf, 64, m0, n0, acc);
    }
    __syncthreads();

    // -- phase 4: mask L with decay; attn = (I - L)^(-1) by forward subst.
    for (int i = tid; i < 4096; i += 256) {
      const int r = i >> 6, c = i & 63;
      Lf[i] = (r > c) ? -Lf[i] * __expf(gcum[r] - gcum[c]) : 0.f;
      attnf[i] = (r == c) ? 1.f : 0.f;
    }
    __syncthreads();
    if (tid < 64) {                                 // columns independent
      const int c = tid;
      for (int r = 1; r < 64; ++r) {
        float s2 = attnf[r * 64 + c];
        for (int j = 0; j < r; ++j) s2 += Lf[r * 64 + j] * attnf[j * 64 + c];
        attnf[r * 64 + c] = s2;
      }
    }
    __syncthreads();
    for (int i = tid; i < 4096; i += 256) attnb[i] = f2bf(attnf[i]);
    __syncthreads();

    // -- phase 5: v_intra = attn@vbb ; k_cumdecay = attn@kgb (64 tiles)
    for (int tt = w; tt < 64; tt += 8) {
      const int which = tt >> 5;
      const int t5 = tt & 31;
      const int m0 = (t5 >> 3) * 16, n0 = (t5 & 7) * 16;
      Acc8 acc; zero_acc(acc);
#pragma unroll
      for (int ks = 0; ks < 2; ++ks) {
        v16bf a = ld_a_rm(attnb, 64, m0, ks * 32);
        v16bf b = ld_b_rm(which ? kgb : vbb, DK, n0, ks * 32);
        acc.v = wmma_bf16(a, b, acc.v);
      }
      if (which) st_tile_bf16(kcdb, DK, m0, n0, acc);
      else       st_tile(vintf, DK, m0, n0, acc);
    }
    __syncthreads();

    // -- phase 6: tmpf = k_cumdecay @ state
    for (int tt = w; tt < 32; tt += 8) {
      const int m0 = (tt >> 3) * 16, n0 = (tt & 7) * 16;
      Acc8 acc; zero_acc(acc);
#pragma unroll
      for (int ks = 0; ks < 4; ++ks) {
        v16bf a = ld_a_rm(kcdb, DK, m0, ks * 32);
        v16bf b = ld_b_rm_f32(state, DV, n0, ks * 32);
        acc.v = wmma_bf16(a, b, acc.v);
      }
      st_tile(tmpf, DV, m0, n0, acc);
    }
    __syncthreads();

    // -- phase 7: v_new = v_intra - tmp ; a_i = mask(araw * decay)
    for (int i = tid; i < CHK * DV; i += 256) vnewb[i] = f2bf(vintf[i] - tmpf[i]);
    for (int i = tid; i < 4096; i += 256) {
      const int r = i >> 6, c = i & 63;
      ab[i] = f2bf((r >= c) ? af[i] * __expf(gcum[r] - gcum[c]) : 0.f);
    }
    __syncthreads();

    // -- phase 8: out = qgb@state + a_i@v_new  -> global
    for (int tt = w; tt < 32; tt += 8) {
      const int m0 = (tt >> 3) * 16, n0 = (tt & 7) * 16;
      Acc8 acc; zero_acc(acc);
#pragma unroll
      for (int ks = 0; ks < 4; ++ks) {
        v16bf a = ld_a_rm(qgb, DK, m0, ks * 32);
        v16bf b = ld_b_rm_f32(state, DV, n0, ks * 32);
        acc.v = wmma_bf16(a, b, acc.v);
      }
#pragma unroll
      for (int ks = 0; ks < 2; ++ks) {
        v16bf a = ld_a_rm(ab, 64, m0, ks * 32);
        v16bf b = ld_b_rm(vnewb, DV, n0, ks * 32);
        acc.v = wmma_bf16(a, b, acc.v);
      }
      const int lane = tid & 31;
      const int n = n0 + (lane & 15), mb = m0 + ((lane >> 4) << 3);
#pragma unroll
      for (int r = 0; r < 8; ++r) oh[(size_t)(t0 + mb + r) * DV + n] = acc.f[r];
    }
    __syncthreads();

    // -- phase 9: state = state*exp(g_last) + (k*exp(gl-g))^T @ v_new
    const float egl = __expf(gl);
    for (int tt = w; tt < 64; tt += 8) {
      const int m0 = (tt >> 3) * 16, n0 = (tt & 7) * 16;
      const int lane = tid & 31;
      const int n = n0 + (lane & 15), mb = m0 + ((lane >> 4) << 3);
      Acc8 acc;
#pragma unroll
      for (int r = 0; r < 8; ++r) acc.f[r] = state[(mb + r) * DV + n] * egl;
#pragma unroll
      for (int ks = 0; ks < 2; ++ks) {
        v16bf a = ld_a_cm(ksb, DK, m0, ks * 32);   // A[m][k] = ksb[t][dk]
        v16bf b = ld_b_rm(vnewb, DV, n0, ks * 32);
        acc.v = wmma_bf16(a, b, acc.v);
      }
#pragma unroll
      for (int r = 0; r < 8; ++r) state[(mb + r) * DV + n] = acc.f[r];
    }
    __syncthreads();
  }
}

// ---------------------------------------------------------------------------
// Gated RMSNorm: x = core * silu(z); x *= rsqrt(mean(x^2)+eps) * w
// One wave32 per (t, head) row of 128 elems; shfl reduction.
// ---------------------------------------------------------------------------
__global__ __launch_bounds__(256) void norm_kernel(const float* __restrict__ core,
                                                   const float* __restrict__ qkvz,
                                                   const float* __restrict__ nw,
                                                   float* __restrict__ outb) {
  const int w = threadIdx.x >> 5, lane = threadIdx.x & 31;
  const int row = blockIdx.x * 8 + w;                 // t*32 + h
  const int t = row >> 5, h = row & 31;
  const size_t zbase = (size_t)t * QKVZ_COLS + (h >> 1) * 768 + 512 + (h & 1) * 128;
  const size_t cbase = (size_t)h * S_LEN * DV + (size_t)t * DV;
  float x[4]; float ss = 0.f;
#pragma unroll
  for (int i = 0; i < 4; ++i) {
    const int d = lane + 32 * i;
    const float z = qkvz[zbase + d];
    const float xv = core[cbase + d] * (z / (1.f + __expf(-z)));
    x[i] = xv; ss += xv * xv;
  }
#pragma unroll
  for (int m = 16; m >= 1; m >>= 1) ss += __shfl_xor(ss, m, 32);
  const float r = rsqrtf(ss * (1.f / 128.f) + 1e-6f);
#pragma unroll
  for (int i = 0; i < 4; ++i) {
    const int d = lane + 32 * i;
    outb[(size_t)t * (NVH * DV) + h * DV + d] = x[i] * r * nw[d];
  }
}

// ---------------------------------------------------------------------------
extern "C" void kernel_launch(void* const* d_in, const int* in_sizes, int n_in,
                              void* d_out, int out_size, void* d_ws, size_t ws_size,
                              hipStream_t stream) {
  const float* hs      = (const float*)d_in[0];
  const float* W_qkvz  = (const float*)d_in[1];
  const float* W_ba    = (const float*)d_in[2];
  const float* convw   = (const float*)d_in[3];
  const float* A_log   = (const float*)d_in[4];
  const float* dt_bias = (const float*)d_in[5];
  const float* nw      = (const float*)d_in[6];
  const float* W_out   = (const float*)d_in[7];
  float* out = (float*)d_out;

  char* ws = (char*)d_ws;
  size_t off = 0;
  float* qkvz = (float*)(ws + off); off += (size_t)S_LEN * QKVZ_COLS * 4;   // 201.3 MB
  float* ba   = (float*)(ws + off); off += (size_t)S_LEN * 64 * 4;
  float* qc   = (float*)(ws + off); off += (size_t)NKH * S_LEN * DK * 4;    // 33.5 MB
  float* kc   = (float*)(ws + off); off += (size_t)NKH * S_LEN * DK * 4;
  float* vc   = (float*)(ws + off); off += (size_t)NVH * S_LEN * DV * 4;    // 67 MB
  float* g    = (float*)(ws + off); off += (size_t)NVH * S_LEN * 4;
  float* beta = (float*)(ws + off); off += (size_t)NVH * S_LEN * 4;
  float* coreo = (float*)(ws + off); off += (size_t)NVH * S_LEN * DV * 4;   // 67 MB
  float* normed = qc;  // qc+kc (67MB) dead after core_kernel; reuse for normed

  const dim3 blk(256);

  // qkvz = hs @ W_qkvz   [4096,2048]x[2048,12288]
  gemm_bf16wmma<<<dim3(QKVZ_COLS / BN, S_LEN / BM), blk, 0, stream>>>(
      hs, W_qkvz, qkvz, S_LEN, QKVZ_COLS, EMB);
  // ba = hs @ W_ba       [4096,2048]x[2048,64]
  gemm_bf16wmma<<<dim3(1, S_LEN / BM), blk, 0, stream>>>(
      hs, W_ba, ba, S_LEN, 64, EMB);
  // depthwise causal conv + SiLU
  conv_silu_kernel<<<(S_LEN * CONV_DIM) / 256, blk, 0, stream>>>(
      qkvz, convw, qc, kc, vc);
  // gates
  gbeta_kernel<<<(S_LEN * NVH) / 256, blk, 0, stream>>>(
      ba, A_log, dt_bias, g, beta);
  // chunked gated delta-rule core: 1 WG per value head, state in 312KB LDS
  core_kernel<<<NVH, blk, CORE_SMEM, stream>>>(qc, kc, vc, g, beta, coreo);
  // gated RMSNorm
  norm_kernel<<<(S_LEN * NVH) / 8, blk, 0, stream>>>(coreo, qkvz, nw, normed);
  // out = normed @ W_out [4096,4096]x[4096,2048]
  gemm_bf16wmma<<<dim3(EMB / BN, S_LEN / BM), blk, 0, stream>>>(
      normed, W_out, out, S_LEN, EMB, NVH * DV);

  (void)in_sizes; (void)n_in; (void)out_size; (void)ws_size;
}
